// ModulatedConv2d_34737695490716
// MI455X (gfx1250) — matrix operand order repaired and checked by reading, not compiled, not run on hardware
//
#include <hip/hip_runtime.h>
#include <hip/hip_bf16.h>
#include <math.h>

#define CIN   512
#define COUT  512
#define SDIM  512
#define HH    64
#define WW    64
#define HWSZ  4096
#define KTAPS 9
#define EPSV  1e-8f
#define SCALEF 0.014731391274719741f  /* 1/sqrt(512*9) */

#define USE_TDM 1   /* set 0 to fall back to manual uint4 weight staging */

typedef unsigned short u16;
typedef __attribute__((ext_vector_type(16))) __bf16 v16bf;
typedef __attribute__((ext_vector_type(8)))  __bf16 v8bf;
typedef __attribute__((ext_vector_type(8)))  float  v8f;
typedef __attribute__((ext_vector_type(4)))  unsigned int u32x4;
typedef __attribute__((ext_vector_type(8)))  int i32x8;

__device__ __forceinline__ u16 f2bf(float f) {
    unsigned u = __float_as_uint(f);
    unsigned r = (u + 0x7fffu + ((u >> 16) & 1u)) >> 16;   // round-to-nearest-even
    return (u16)r;
}

// ---------------- s[n,c] = style[n,:] . style_w[c,:] + style_b[c] ----------------
__global__ void style_kernel(const float* __restrict__ style,
                             const float* __restrict__ style_w,
                             const float* __restrict__ style_b,
                             float* __restrict__ s, float* __restrict__ s2) {
    int id = blockIdx.x * blockDim.x + threadIdx.x;        // 8*512 = 4096
    int n = id >> 9, c = id & 511;
    const float* st = style + (size_t)n * SDIM;
    const float* wr = style_w + (size_t)c * SDIM;
    float acc = 0.f;
    for (int d = 0; d < SDIM; ++d) acc = fmaf(st[d], wr[d], acc);
    acc += style_b[c];
    s[id] = acc;
    s2[id] = acc * acc;
}

// ---------------- wsq[o,i] = sum_t weight[o,i,t]^2 ----------------
__global__ void wsq_kernel(const float* __restrict__ w, float* __restrict__ wsq) {
    int id = blockIdx.x * blockDim.x + threadIdx.x;        // 512*512
    const float* wp = w + (size_t)id * KTAPS;
    float a = 0.f;
    for (int t = 0; t < KTAPS; ++t) { float v = wp[t]; a = fmaf(v, v, a); }
    wsq[id] = a;
}

// ---------------- demod[n,o] = rsqrt(SCALE^2 * sum_i s2[n,i]*wsq[o,i] + eps) ----------------
__global__ void demod_kernel(const float* __restrict__ s2, const float* __restrict__ wsq,
                             float* __restrict__ dm) {
    int id = blockIdx.x * blockDim.x + threadIdx.x;        // 8*512
    int n = id >> 9, o = id & 511;
    const float* a = s2 + (size_t)n * CIN;
    const float* b = wsq + (size_t)o * CIN;
    float acc = 0.f;
    for (int i = 0; i < CIN; ++i) acc = fmaf(a[i], b[i], acc);
    dm[id] = rsqrtf(SCALEF * SCALEF * acc + EPSV);
}

// ---------------- Amod[n][t][o][i] = weight[o,i,t]*SCALE*s[n,i]*demod[n,o]  (bf16) ----------------
__global__ void modw_kernel(const float* __restrict__ w, const float* __restrict__ s,
                            const float* __restrict__ dm, u16* __restrict__ Amod) {
    size_t id = (size_t)blockIdx.x * blockDim.x + threadIdx.x;   // 8*9*512*512
    int i = (int)(id & 511);
    int o = (int)((id >> 9) & 511);
    int t = (int)((id >> 18) % 9);
    int n = (int)(id / ((size_t)9 << 18));
    float v = w[((size_t)o * CIN + i) * KTAPS + t] * SCALEF
            * s[(size_t)n * CIN + i] * dm[(size_t)n * COUT + o];
    Amod[id] = f2bf(v);
}

// ---------------- main conv: tap-split implicit GEMM with bf16 WMMA + TDM staging ----------
// block = 256 threads (8 waves). Tile: M=128 (O) x N=128 (pixels = 2 rows of W=64).
// Per kc (32-channel chunk): TDM DMAs all 9 taps' weight tiles into LDS (hw padded rows)
// while the VALU stages the haloed x tile; 2 barriers per kc; 72 WMMAs/wave per kc.
__global__ __launch_bounds__(256)
void conv_wmma_kernel(const float* __restrict__ x, const u16* __restrict__ Amod,
                      float* __restrict__ out) {
    __shared__ u16 ldsX[4 * 66 * 40];        // 21120 B: [row 0..3][col 0..65][ch+pad]
    __shared__ u16 ldsA9[KTAPS * 128 * 40];  // 92160 B: 9 taps x [o_local][k+pad]

    const int tid   = threadIdx.x;
    const int lane  = tid & 31;
    const int wv    = tid >> 5;
    const int mquad = wv & 3;           // M offset = mquad*32
    const int nhalf = wv >> 2;          // N offset = nhalf*64
    const int lm    = lane & 15;
    const int hi    = lane >> 4;        // 0 or 1

    const int n     = blockIdx.z;
    const int oBase = blockIdx.y * 128;
    const int pBase = blockIdx.x * 128;
    const int y0    = pBase >> 6;       // first of the 2 output rows

#if USE_TDM
    const unsigned ldsABase = (unsigned)(size_t)(&ldsA9[0]);  // LDS byte offset
#endif

    // zero the haloed x tile (border cols / OOB rows stay zero forever)
    for (int idx = tid; idx < 4 * 66 * 40; idx += 256) ldsX[idx] = 0;
    __syncthreads();

    v8f acc[2][4];
#pragma unroll
    for (int a = 0; a < 2; ++a)
#pragma unroll
        for (int b = 0; b < 4; ++b)
#pragma unroll
            for (int r = 0; r < 8; ++r) acc[a][b][r] = 0.f;

    union V16 { v16bf v; v8bf h[2]; };

    for (int kc = 0; kc < CIN / 32; ++kc) {
#if USE_TDM
        // ---- wave 0: issue 9 TDM tile loads (2D, 128 rows x 64B, hw-padded +16B/row) ----
        if (wv == 0) {
#pragma unroll
            for (int t = 0; t < KTAPS; ++t) {
                unsigned long long gaddr = (unsigned long long)(size_t)(Amod
                    + ((size_t)(n * KTAPS + t) * COUT + oBase) * CIN + kc * 32);
                u32x4 g0;
                g0.x = 1u;                                    // count=1, user mode
                g0.y = ldsABase + (unsigned)(t * 10240);      // lds_addr (bytes)
                g0.z = (unsigned)gaddr;                       // global_addr[31:0]
                g0.w = (unsigned)(gaddr >> 32) | (2u << 30);  // addr[56:32] | type=2
                i32x8 g1 = {
                    (int)((1u << 16) | (1u << 20) | (3u << 22) | (3u << 25)),
                    // data_size=2B, pad_enable, pad_interval=16 dw, pad_amount=4 dw
                    (int)0xFFFF0000u,          // tensor_dim0 lo16 (dim0=0x7FFFFFFF)
                    (int)0xFFFF7FFFu,          // dim0 hi16 | tensor_dim1 lo16
                    (int)((32u << 16) | 0x7FFFu), // dim1 hi16 | tile_dim0=32
                    (int)128,                  // tile_dim1=128, tile_dim2=0
                    (int)CIN,                  // tensor_dim0_stride = 512 elements
                    0, 0
                };
                asm volatile("tensor_load_to_lds %0, %1"
                             :: "s"(g0), "s"(g1) : "memory");
            }
        }
#else
        // ---- manual weight staging: 9 taps x 128 rows x 32 halves ----
#pragma unroll
        for (int t = 0; t < KTAPS; ++t) {
#pragma unroll
            for (int cc = 0; cc < 2; ++cc) {
                const int c    = tid + cc * 256;     // 512 16B chunks per tap
                const int ol   = c >> 2;
                const int off8 = (c & 3) << 3;
                const uint4 v = *(const uint4*)&Amod[
                    ((size_t)(n * KTAPS + t) * COUT + oBase + ol) * CIN + kc * 32 + off8];
                *(uint4*)&ldsA9[t * 5120 + ol * 40 + off8] = v;
            }
        }
#endif

        // ---- all waves: stage x tile (4 rows x 64 cols x 32 ch), f32 -> bf16 ----
        {
            const int cl = tid >> 3;            // channel 0..31
            const int x0 = (tid & 7) << 3;      // 8 consecutive cols
            for (int r = 0; r < 4; ++r) {
                const int gy = y0 - 1 + r;
                if (gy >= 0 && gy < HH) {        // uniform per block
                    const float* src =
                        x + (((size_t)(n * CIN + kc * 32 + cl) * HH + gy) * WW + x0);
                    float4 f0 = *(const float4*)src;
                    float4 f1 = *(const float4*)(src + 4);
                    int base = (r * 66 + 1 + x0) * 40 + cl;
                    ldsX[base      ] = f2bf(f0.x);
                    ldsX[base + 40 ] = f2bf(f0.y);
                    ldsX[base + 80 ] = f2bf(f0.z);
                    ldsX[base + 120] = f2bf(f0.w);
                    ldsX[base + 160] = f2bf(f1.x);
                    ldsX[base + 200] = f2bf(f1.y);
                    ldsX[base + 240] = f2bf(f1.z);
                    ldsX[base + 280] = f2bf(f1.w);
                }
            }
        }

#if USE_TDM
        if (wv == 0) asm volatile("s_wait_tensorcnt 0x0" ::: "memory");
#endif
        __syncthreads();

        // ---- compute: 9 taps x (2 M-tiles x 4 N-tiles) WMMAs ----
#pragma unroll
        for (int t = 0; t < KTAPS; ++t) {
            const int dy = t / 3 - 1, dx = t % 3 - 1;

            // A fragments: lane<16 -> K 0..7 & 16..23 ; lane>=16 -> K 8..15 & 24..31
            V16 afr[2];
#pragma unroll
            for (int mti = 0; mti < 2; ++mti) {
                const int off = t * 5120 + (mquad * 32 + mti * 16 + lm) * 40 + hi * 8;
                afr[mti].h[0] = *(const v8bf*)&ldsA9[off];
                afr[mti].h[1] = *(const v8bf*)&ldsA9[off + 16];
            }

            // B fragments: column = pixel; lane<16 -> K 0..7 & 16..23 ; lane>=16 -> K 8..15 & 24..31
#pragma unroll
            for (int nti = 0; nti < 4; ++nti) {
                const int pl   = nhalf * 64 + nti * 16 + lm;    // 0..127
                const int lrow = (pl >> 6) + dy + 1;            // 0..3
                const int lcol = (pl & 63) + dx + 1;            // 0..65
                const int off  = (lrow * 66 + lcol) * 40 + hi * 8;
                V16 bfr;
                bfr.h[0] = *(const v8bf*)&ldsX[off];
                bfr.h[1] = *(const v8bf*)&ldsX[off + 16];
#pragma unroll
                for (int mti = 0; mti < 2; ++mti) {
                    acc[mti][nti] = __builtin_amdgcn_wmma_f32_16x16x32_bf16(
                        false, afr[mti].v, false, bfr.v,
                        (short)0, acc[mti][nti], false, false);
                }
            }
        }
        __syncthreads();   // protect LDS before next chunk's staging
    }

    // epilogue: C layout -> vgpr r, lane: M = r + 8*hi, N = lane%16
#pragma unroll
    for (int mti = 0; mti < 2; ++mti) {
#pragma unroll
        for (int nti = 0; nti < 4; ++nti) {
            const int o = oBase + mquad * 32 + mti * 16 + hi * 8;
            const int p = pBase + nhalf * 64 + nti * 16 + lm;
            float* op = out + ((size_t)(n * COUT + o) * HWSZ + p);
            v8f c = acc[mti][nti];
#pragma unroll
            for (int r = 0; r < 8; ++r) op[(size_t)r * HWSZ] = c[r];
        }
    }
}

extern "C" void kernel_launch(void* const* d_in, const int* in_sizes, int n_in,
                              void* d_out, int out_size, void* d_ws, size_t ws_size,
                              hipStream_t stream) {
    const float* x       = (const float*)d_in[0];
    const float* style   = (const float*)d_in[1];
    const float* weight  = (const float*)d_in[2];
    const float* style_w = (const float*)d_in[3];
    const float* style_b = (const float*)d_in[4];
    float* out = (float*)d_out;

    char* ws = (char*)d_ws;
    float* s    = (float*)(ws);                       // 16 KB
    float* s2   = (float*)(ws + 16384);               // 16 KB
    float* wsq  = (float*)(ws + 32768);               // 1 MB
    float* dm   = (float*)(ws + 32768 + 1048576);     // 16 KB
    u16*   Amod = (u16*)  (ws + (size_t)(2u << 20));  // 37.75 MB bf16 modulated weights

    style_kernel<<<16,    256, 0, stream>>>(style, style_w, style_b, s, s2);
    wsq_kernel  <<<1024,  256, 0, stream>>>(weight, wsq);
    demod_kernel<<<16,    256, 0, stream>>>(s2, wsq, dm);
    modw_kernel <<<73728, 256, 0, stream>>>(weight, s, dm, Amod);
    conv_wmma_kernel<<<dim3(32, 4, 8), 256, 0, stream>>>(x, Amod, out);
}
